// MSR_13606456394497
// MI455X (gfx1250) — matrix operand order repaired and checked by reading, not compiled, not run on hardware
//
#include <hip/hip_runtime.h>
#include <hip/hip_bf16.h>

#define S_LEN 2048
#define HID   2048
#define H3    6144
#define NH    32
#define HD    64

typedef __attribute__((ext_vector_type(16))) __bf16 v16bf;
typedef __attribute__((ext_vector_type(8)))  float  v8f;
typedef __attribute__((ext_vector_type(4)))  unsigned int v4u;
typedef __attribute__((ext_vector_type(8)))  int v8i;
typedef __attribute__((ext_vector_type(4)))  int v4i;

#define WMMA_BF16(a, b, c) \
  __builtin_amdgcn_wmma_f32_16x16x32_bf16(false, (a), false, (b), (short)0, (c), false, false)

#if defined(__has_builtin)
#if __has_builtin(__builtin_amdgcn_tensor_load_to_lds) && \
    __has_builtin(__builtin_amdgcn_s_wait_tensorcnt)
#define USE_TDM 1
#endif
#endif

// ---------------------------------------------------------------------------
// A fragment 16x32 (MxK) from row-major source, stride lda, in this code's own k order (B packed to match):
// lanes 0-15: M=lane, K 0-15; lanes 16-31: M=lane-16, K 16-31 (contiguous).
__device__ inline v16bf frag_a(const __bf16* a, int lda) {
  const int lane = threadIdx.x & 31;
  const int m  = lane & 15;
  const int kh = (lane >> 4) << 4;
  v16bf f;
#pragma unroll
  for (int g = 0; g < 2; ++g)
#pragma unroll
    for (int p = 0; p < 8; ++p)
      f[g * 8 + p] = a[m * lda + g * 8 + kh + p];
  return f;
}

// B fragment from fragment-native packed block: 32 lanes x 16 contiguous bf16.
__device__ inline v16bf frag_packed(const __bf16* block) {
  return *(const v16bf*)(block + (threadIdx.x & 31) * 16);
}

// packed-block index helpers -------------------------------------------------
__device__ inline size_t kpack_idx(int h, int d, int s) {
  return ((((size_t)h * (S_LEN / 16) + (s >> 4)) * 2 + (d >> 5)) * 32
          + (((d >> 4) & 1) * 16 + (s & 15))) * 16 + (d & 15);
}
__device__ inline size_t vpack_idx(int h, int s, int d) {
  return ((((size_t)h * (S_LEN / 32) + (s >> 5)) * 4 + (d >> 4)) * 32
          + (((s >> 4) & 1) * 16 + (d & 15))) * 16 + (s & 15);
}

#ifdef USE_TDM
// TDM 2D tile load: data_size=8B, tile = tile0 (8B units) x tile1 rows,
// row stride = stride0 (8B units). D# per CDNA5 ISA 8.3/8.4.
// clang-23 / therock-10.0 arity: (v4u, v8i, v4i, v4i, v8i, i32 cpol)
__device__ inline void tdm_load_2d(unsigned int lds_addr, unsigned long long gaddr,
                                   unsigned int tile0, unsigned int tile1,
                                   unsigned int stride0) {
  v4u g0;
  g0[0] = 1u;                                   // count=1, user descriptor
  g0[1] = lds_addr;                             // LDS byte address
  g0[2] = (unsigned int)gaddr;                  // global_addr[31:0]
  g0[3] = (unsigned int)((gaddr >> 32) & 0x1FFFFFFu) | (2u << 30);  // [56:32]|type=2
  const unsigned int td0 = 1u << 30, td1 = 1u << 30;  // huge tensor dims: no OOB clip
  v8i g1;
  g1[0] = (int)(3u << 16);                      // workgroup_mask=0, data_size=8B
  g1[1] = (int)((td0 & 0xFFFFu) << 16);         // barrier_addr=0 | tensor_dim0 lo
  g1[2] = (int)(((td0 >> 16) & 0xFFFFu) | ((td1 & 0xFFFFu) << 16));
  g1[3] = (int)(((td1 >> 16) & 0xFFFFu) | (tile0 << 16));   // | tile_dim0
  g1[4] = (int)(tile1 & 0xFFFFu);               // tile_dim1 | tile_dim2=0
  g1[5] = (int)stride0;                         // tensor_dim0_stride[31:0]
  g1[6] = 0;                                    // stride0[47:32] | stride1 lo
  g1[7] = 0;
  v4i z4 = {};
  v8i z8 = {};
  __builtin_amdgcn_tensor_load_to_lds(g0, g1, z4, z4, z8, 0);
}
#endif

// ---------------------------------------------------------------------------
// Generic bf16 GEMM: C[M,N] = A[M,K] * Bp, A bf16 row-major, Bp packed
// fragment-native ([K/32][N/16][32][16]); out fp32 (Cf) or bf16 (Cb).
// Workgroup tile 64x128, 8 waves, K step 32, double-buffered TDM staging.
__global__ __launch_bounds__(256) void gemm_bf16_kernel(
    const __bf16* __restrict__ A, const __bf16* __restrict__ Bp,
    float* __restrict__ Cf, __bf16* __restrict__ Cb, int M, int N, int K) {
  __shared__ __align__(32) __bf16 As[2][64 * 32];
  __shared__ __align__(32) __bf16 Bs[2][32 * 128];   // 8 packed blocks * 512
  const int t = threadIdx.x, lane = t & 31, wave = t >> 5;
  const int mBase = blockIdx.y * 64, nBase = blockIdx.x * 128;
  const int mw = (wave & 3) * 16, nw = (wave >> 2) * 64;
  const int nb16 = N >> 4;
  v8f acc[4] = {};

#ifdef USE_TDM
  // Async engine copy: wave 0 issues one 2D + one 1D TDM descriptor per stage;
  // TENSORcnt-tracked, overlapped with WMMA on the other LDS buffer.
  auto issue = [&](int k0, int buf) {
    if (wave != 0) return;
    tdm_load_2d((unsigned int)(uintptr_t)&As[buf][0],
                (unsigned long long)(uintptr_t)(A + (size_t)mBase * K + k0),
                /*tile0=*/8, /*tile1=*/64, /*stride0=*/(unsigned int)(K >> 2));
    const __bf16* Bg = Bp + (((size_t)(k0 >> 5) * nb16 + (nBase >> 4)) << 9);
    tdm_load_2d((unsigned int)(uintptr_t)&Bs[buf][0],
                (unsigned long long)(uintptr_t)Bg,
                /*tile0=*/1024, /*tile1=*/1, /*stride0=*/1024);
  };
  issue(0, 0);
  for (int k0 = 0; k0 < K; k0 += 32) {
    const int cur = (k0 >> 5) & 1;
    if (wave == 0) __builtin_amdgcn_s_wait_tensorcnt((short)0);
    __syncthreads();                       // tile `cur` visible to all waves
    if (k0 + 32 < K) issue(k0 + 32, cur ^ 1);
    v16bf a = frag_a(&As[cur][0] + mw * 32, 32);
#pragma unroll
    for (int tt = 0; tt < 4; ++tt) {
      v16bf b = frag_packed(&Bs[cur][0] + (((nw >> 4) + tt) << 9));
      acc[tt] = WMMA_BF16(a, b, acc[tt]);
    }
  }
#else
  const int arow = t >> 2, acol = (t & 3) * 8;
  for (int k0 = 0; k0 < K; k0 += 32) {
    uint4 av = *(const uint4*)(A + (size_t)(mBase + arow) * K + k0 + acol);
    const __bf16* Bg = Bp + (((size_t)(k0 >> 5) * nb16 + (nBase >> 4)) << 9);
    v16bf bv = *(const v16bf*)(Bg + t * 16);
    if (k0 + 32 < K) {
      __builtin_prefetch(A + (size_t)(mBase + arow) * K + k0 + 32 + acol, 0, 1);
      __builtin_prefetch(Bg + ((size_t)nb16 << 9) + t * 16, 0, 1);
    }
    __syncthreads();
    *(uint4*)(&As[0][0] + arow * 32 + acol) = av;
    *(v16bf*)(&Bs[0][0] + t * 16) = bv;
    __syncthreads();
    v16bf a = frag_a(&As[0][0] + mw * 32, 32);
#pragma unroll
    for (int tt = 0; tt < 4; ++tt) {
      v16bf b = frag_packed(&Bs[0][0] + (((nw >> 4) + tt) << 9));
      acc[tt] = WMMA_BF16(a, b, acc[tt]);
    }
  }
#endif

#pragma unroll
  for (int tt = 0; tt < 4; ++tt)
#pragma unroll
    for (int r = 0; r < 8; ++r) {
      const int row = mBase + mw + r + ((lane >> 4) << 3);
      const int col = nBase + nw + tt * 16 + (lane & 15);
      const float v = acc[tt][r];
      if (Cb) Cb[(size_t)row * N + col] = (__bf16)v;
      else    Cf[(size_t)row * N + col] = v;
    }
}

// ---------------------------------------------------------------------------
__global__ __launch_bounds__(256) void conv_bf16_kernel(const float* __restrict__ in,
                                                        __bf16* __restrict__ out, int n) {
  int i = blockIdx.x * 256 + threadIdx.x;
  if (i < n) out[i] = (__bf16)in[i];
}

// Pack W[N][K] (f32) into fragment-native B = W^T blocks (bf16)
__global__ __launch_bounds__(256) void pack_b_kernel(const float* __restrict__ in,
                                                     __bf16* __restrict__ out,
                                                     int N, int K) {
  int i = blockIdx.x * 256 + threadIdx.x;
  if (i >= N * K) return;
  const int e = i & 15;
  const int lane = (i >> 4) & 31;
  const int blk = i >> 9;
  const int nb16 = N >> 4;
  const int bk = blk / nb16, bn = blk - bk * nb16;
  const int k = (bk << 5) + ((lane >> 4) << 4) + e;
  const int n = (bn << 4) + (lane & 15);
  out[i] = (__bf16)in[(size_t)n * K + k];
}

// ---------------------------------------------------------------------------
// xPos rotary; writes q[h][s][d] (A-operand layout), kPack, vPack (bf16)
__global__ __launch_bounds__(256) void rotary_kernel(const __bf16* __restrict__ qkv,
                                                     __bf16* __restrict__ qh,
                                                     __bf16* __restrict__ kP,
                                                     __bf16* __restrict__ vP) {
  const int idx = blockIdx.x * 256 + threadIdx.x;
  const int h = idx >> 11;
  const int s = idx & 2047;
  const __bf16* row = qkv + (size_t)s * H3 + h * 192;
  const float power = ((float)s - 1024.0f) * (1.0f / 512.0f);
  const float L2T = 13.28771237954945f;                 // log2(10000)
#pragma unroll
  for (int p = 0; p < 16; ++p) {
    const int d0 = 2 * p, d1 = d0 + 1;
    const float fr = (float)s * exp2f(-((float)d0 * (1.0f / 32.0f)) * L2T);
    const float cw = cosf(fr), sw = sinf(fr);
    const float e0 = power * log2f((2.0f * (float)(d0 & 15) + 12.8f) * (1.0f / 44.8f));
    const float e1 = power * log2f((2.0f * (float)(d1 & 15) + 12.8f) * (1.0f / 44.8f));
    const float q0 = (float)row[d0],      q1 = (float)row[d1];
    const float k0 = (float)row[64 + d0], k1 = (float)row[64 + d1];
    qh[((size_t)h * S_LEN + s) * HD + d0] = (__bf16)((q0 * cw - q1 * sw) * exp2f(e0));
    qh[((size_t)h * S_LEN + s) * HD + d1] = (__bf16)((q1 * cw + q0 * sw) * exp2f(e1));
    kP[kpack_idx(h, d0, s)] = (__bf16)((k0 * cw - k1 * sw) * exp2f(-e0));
    kP[kpack_idx(h, d1, s)] = (__bf16)((k1 * cw + k0 * sw) * exp2f(-e1));
  }
  for (int d = 32; d < 64; ++d) {
    qh[((size_t)h * S_LEN + s) * HD + d] = row[d];
    kP[kpack_idx(h, d, s)] = row[64 + d];
  }
  for (int d = 0; d < 64; ++d)
    vP[vpack_idx(h, s, d)] = row[128 + d];
}

// ---------------------------------------------------------------------------
// Retention: flash-style with per-tile decay (uniform trip count per block)
__global__ __launch_bounds__(128) void retention_kernel(const __bf16* __restrict__ q,
                                                        const __bf16* __restrict__ kP,
                                                        const __bf16* __restrict__ vP,
                                                        float* __restrict__ ret) {
  __shared__ __align__(32) __bf16 sS[4][16 * 32];
  const int h = blockIdx.y;
  const int blockRow = blockIdx.x * 64;
  const int wave = threadIdx.x >> 5, lane = threadIdx.x & 31;
  const int nlo = lane & 15, mhi = (lane >> 4) << 3;
  const int iBase = blockRow + wave * 16;
  const float gamma = 1.0f - exp2f(-5.0f - (float)h);
  const float l2g = log2f(gamma);

  const __bf16* qp = q + ((size_t)h * S_LEN + iBase) * HD;
  const v16bf aq0 = frag_a(qp, HD);
  const v16bf aq1 = frag_a(qp + 32, HD);
  v8f acc[4] = {};

  const int i0 = iBase + mhi;
  const int jMax = blockRow + 63;
  for (int j0 = 0; j0 <= jMax; j0 += 32) {
    __syncthreads();
#pragma unroll
    for (int sub = 0; sub < 2; ++sub) {
      const int jb = j0 + sub * 16;
      const __bf16* kb = kP + ((size_t)(h * (S_LEN / 16) + (jb >> 4)) << 10);
      v8f c = {};
      c = WMMA_BF16(aq0, frag_packed(kb), c);
      c = WMMA_BF16(aq1, frag_packed(kb + 512), c);
      const int j = jb + nlo;
      const float base = exp2f((float)(i0 - j) * l2g);   // gamma^(i0-j)
      float gp = 1.0f;
#pragma unroll
      for (int r = 0; r < 8; ++r) {
        const int i = i0 + r;
        const float dec = (i >= j) ? base * gp : 0.0f;
        sS[wave][(mhi + r) * 32 + sub * 16 + nlo] = (__bf16)(c[r] * dec);
        gp *= gamma;
      }
    }
    __syncthreads();
    const v16bf aS = frag_a(sS[wave], 32);
    const __bf16* vb = vP + ((size_t)(h * (S_LEN / 32) + (j0 >> 5)) << 11);
#pragma unroll
    for (int tt = 0; tt < 4; ++tt)
      acc[tt] = WMMA_BF16(aS, frag_packed(vb + (tt << 9)), acc[tt]);
  }
#pragma unroll
  for (int tt = 0; tt < 4; ++tt)
#pragma unroll
    for (int r = 0; r < 8; ++r) {
      const int i = iBase + r + mhi;
      ret[((size_t)h * S_LEN + i) * HD + tt * 16 + nlo] = acc[tt][r];
    }
}

// ---------------------------------------------------------------------------
__global__ __launch_bounds__(256) void stats_kernel(const float* __restrict__ ret,
                                                    float* __restrict__ st) {
  __shared__ float ssum[256], ssq[256];
  const int h = blockIdx.x, tid = threadIdx.x;
  const float* p = ret + (size_t)h * S_LEN * HD;
  float s = 0.0f, q = 0.0f;
  for (int i = tid; i < S_LEN * HD; i += 256) {
    const float v = p[i];
    s += v; q += v * v;
  }
  ssum[tid] = s; ssq[tid] = q;
  __syncthreads();
  for (int o = 128; o > 0; o >>= 1) {
    if (tid < o) { ssum[tid] += ssum[tid + o]; ssq[tid] += ssq[tid + o]; }
    __syncthreads();
  }
  if (tid == 0) {
    const float inv = 1.0f / (float)(S_LEN * HD);
    const float mean = ssum[0] * inv;
    const float var = ssq[0] * inv - mean * mean;
    st[h * 2] = mean;
    st[h * 2 + 1] = rsqrtf(var + 1e-5f);
  }
}

// ---------------------------------------------------------------------------
__global__ __launch_bounds__(256) void gate_kernel(const float* __restrict__ g,
                                                   const float* __restrict__ ret,
                                                   const float* __restrict__ st,
                                                   const float* __restrict__ gnw,
                                                   const float* __restrict__ gnb,
                                                   __bf16* __restrict__ z) {
  const int idx = blockIdx.x * 256 + threadIdx.x;
  if (idx >= S_LEN * HID) return;
  const int s = idx >> 11;
  const int hid = idx & 2047;
  const int h = hid >> 6, d = hid & 63;
  const float mean = st[h * 2], rstd = st[h * 2 + 1];
  const float rv = ret[((size_t)h * S_LEN + s) * HD + d];
  const float y = (rv - mean) * rstd * gnw[h] + gnb[h];
  const float gg = g[idx];
  const float sp = (gg > 20.0f) ? gg : log1pf(expf(gg));
  const float mish = gg * tanhf(sp);
  z[idx] = (__bf16)(mish * y);
}

// ---------------------------------------------------------------------------
extern "C" void kernel_launch(void* const* d_in, const int* in_sizes, int n_in,
                              void* d_out, int out_size, void* d_ws, size_t ws_size,
                              hipStream_t stream) {
  const float* x    = (const float*)d_in[0];
  const float* wqkv = (const float*)d_in[1];
  const float* wg   = (const float*)d_in[2];
  const float* wo   = (const float*)d_in[3];
  const float* gnw  = (const float*)d_in[4];
  const float* gnb  = (const float*)d_in[5];
  float* out = (float*)d_out;

  char* ws = (char*)d_ws;
  size_t off = 0;
  auto alloc = [&](size_t bytes) -> void* {
    void* p = ws + off;
    off += (bytes + 255) & ~(size_t)255;
    return p;
  };
  __bf16* xb    = (__bf16*)alloc((size_t)S_LEN * HID * 2);
  __bf16* wqkvP = (__bf16*)alloc((size_t)HID * H3 * 2);
  __bf16* wgP   = (__bf16*)alloc((size_t)HID * HID * 2);
  __bf16* woP   = (__bf16*)alloc((size_t)HID * HID * 2);
  __bf16* qkvb  = (__bf16*)alloc((size_t)S_LEN * H3 * 2);
  __bf16* qh    = (__bf16*)alloc((size_t)NH * S_LEN * HD * 2);
  __bf16* kP    = (__bf16*)alloc((size_t)NH * S_LEN * HD * 2);
  __bf16* vP    = (__bf16*)alloc((size_t)NH * S_LEN * HD * 2);
  float*  retf  = (float*)alloc((size_t)NH * S_LEN * HD * 4);
  float*  st    = (float*)alloc(NH * 2 * 4);
  float*  gf    = (float*)alloc((size_t)S_LEN * HID * 4);
  __bf16* zb    = (__bf16*)alloc((size_t)S_LEN * HID * 2);
  (void)ws_size; (void)in_sizes; (void)n_in; (void)out_size;

  conv_bf16_kernel<<<(S_LEN * HID) / 256, 256, 0, stream>>>(x, xb, S_LEN * HID);
  pack_b_kernel<<<(H3 * HID) / 256, 256, 0, stream>>>(wqkv, wqkvP, H3, HID);
  pack_b_kernel<<<(HID * HID) / 256, 256, 0, stream>>>(wg, wgP, HID, HID);
  pack_b_kernel<<<(HID * HID) / 256, 256, 0, stream>>>(wo, woP, HID, HID);

  gemm_bf16_kernel<<<dim3(H3 / 128, S_LEN / 64), 256, 0, stream>>>(
      xb, wqkvP, nullptr, qkvb, S_LEN, H3, HID);

  rotary_kernel<<<(NH * S_LEN) / 256, 256, 0, stream>>>(qkvb, qh, kP, vP);

  retention_kernel<<<dim3(S_LEN / 64, NH), 128, 0, stream>>>(qh, kP, vP, retf);

  stats_kernel<<<NH, 256, 0, stream>>>(retf, st);

  gemm_bf16_kernel<<<dim3(HID / 128, S_LEN / 64), 256, 0, stream>>>(
      xb, wgP, gf, nullptr, S_LEN, HID, HID);

  gate_kernel<<<(S_LEN * HID) / 256, 256, 0, stream>>>(gf, retf, st, gnw, gnb, zb);

  gemm_bf16_kernel<<<dim3(HID / 128, S_LEN / 64), 256, 0, stream>>>(
      zb, woP, out, nullptr, S_LEN, HID, HID);
}
